// S4Compact_9972914062073
// MI455X (gfx1250) — matrix operand order repaired and checked by reading, not compiled, not run on hardware
//
#include <hip/hip_runtime.h>
#include <hip/hip_fp16.h>

// ---------------------------------------------------------------------------
// S4D stack for MI455X (gfx1250, wave32).
//  * S4D long conv  -> chunked parallel scan (3 kernels), NOT FFT.
//  * All 1x1 convs  -> f16 WMMA GEMMs (v_wmma_f32_16x16x32_f16), 32 rows per
//    wave so each LDS B-fragment feeds two WMMAs.
//  * Weight staging global->LDS via Tensor Data Mover (tensor_load_to_lds,
//    TENSORcnt) when the builtin is available; cooperative copy otherwise.
// ---------------------------------------------------------------------------

typedef __attribute__((ext_vector_type(16))) _Float16 v16h;
typedef __attribute__((ext_vector_type(8)))  _Float16 v8h;
typedef __attribute__((ext_vector_type(8)))  float    v8f;

#if !defined(USE_TDM)
#  if defined(__has_builtin)
#    if __has_builtin(__builtin_amdgcn_tensor_load_to_lds)
#      define USE_TDM 1
#    endif
#  endif
#endif
#ifndef USE_TDM
#define USE_TDM 0
#endif

namespace {
constexpr int BB   = 8;
constexpr int LLEN = 32768;
constexpr int DD   = 64;
constexpr int NS   = 16;          // SSM states per channel
constexpr int NLAY = 4;
constexpr int MM   = BB * LLEN;   // 262144 positions
constexpr int NCH  = 64;          // scan chunks per row
constexpr int CHL  = LLEN / NCH;  // 512 steps per chunk
}

__device__ __forceinline__ float gelu_f(float x) {
  float x3 = x * x * x;
  return 0.5f * x * (1.f + tanhf(0.7978845608f * (x + 0.044715f * x3)));
}
__device__ __forceinline__ float sigmoid_f(float x) {
  return 1.f / (1.f + __expf(-x));
}

#if USE_TDM
typedef __attribute__((ext_vector_type(4))) unsigned int u32x4;
typedef __attribute__((ext_vector_type(8))) int          i32x8;
typedef __attribute__((ext_vector_type(4))) int          i32x4;

// 1-D tile DMA global->LDS via the Tensor Data Mover. D# per CDNA5 ISA §8:
// group0: count=1 | lds_addr | 57b global_addr | type=2("image")
// group1: data_size=4B, tensor_dim0 = tile_dim0 = ndwords, dim1 = 1.
__device__ __forceinline__ void tdm_load_tile(const void* gsrc, void* ldst,
                                              unsigned ndwords) {
  unsigned long long ga = (unsigned long long)(size_t)gsrc;
  unsigned lo = (unsigned)(size_t)ldst;  // LDS aperture: addr[31:0] = LDS off
  u32x4 g0;
  g0[0] = 1u;                                              // count = 1
  g0[1] = lo;                                              // lds_addr
  g0[2] = (unsigned)ga;                                    // global_addr lo
  g0[3] = (unsigned)((ga >> 32) & 0x01FFFFFFu) | (2u << 30);  // hi + type
  i32x8 g1;
  g1[0] = 0x20000;                                  // data_size = 2 (4 bytes)
  g1[1] = (int)((ndwords & 0xFFFFu) << 16);         // tensor_dim0[15:0]
  g1[2] = (int)((ndwords >> 16) | (1u << 16));      // dim0 hi | tensor_dim1=1
  g1[3] = (int)((ndwords & 0xFFFFu) << 16);         // tile_dim0
  g1[4] = 1;                                        // tile_dim1 = 1
  g1[5] = (int)ndwords;                             // tensor_dim0_stride
  g1[6] = 0;
  g1[7] = 0;
  i32x4 z4 = {0, 0, 0, 0};
#if __clang_major__ >= 23
  i32x8 z8 = {0, 0, 0, 0, 0, 0, 0, 0};
  __builtin_amdgcn_tensor_load_to_lds(g0, g1, z4, z4, z8, 0);
#else
  __builtin_amdgcn_tensor_load_to_lds(g0, g1, z4, z4, 0);
#endif
}
#endif  // USE_TDM

// ---------------------------------------------------------------------------
// fp32 -> fp16 weight conversion (weights are tiny; done once per launch)
// ---------------------------------------------------------------------------
__global__ void cvt_f32_f16_kernel(const float* __restrict__ s,
                                   _Float16* __restrict__ d, int n) {
  int i = blockIdx.x * 256 + threadIdx.x;
  if (i < n) d[i] = (_Float16)s[i];
}

// ---------------------------------------------------------------------------
// init conv: x[pos,d] = relu(w_init[d]*u[pos] + b_init[d])
// ---------------------------------------------------------------------------
__global__ void init_kernel(const float* __restrict__ u,
                            const float* __restrict__ w,
                            const float* __restrict__ b,
                            float* __restrict__ x) {
  int idx = blockIdx.x * 256 + threadIdx.x;   // MM*DD threads
  int d   = idx & (DD - 1);
  int pos = idx >> 6;
  float v = fmaf(w[d], u[pos], b[d]);
  x[idx] = v > 0.f ? v : 0.f;
}

// ---------------------------------------------------------------------------
// LayerNorm over D per position. One thread per position, two float4 passes.
// ---------------------------------------------------------------------------
template <typename OT>
__global__ void ln_kernel(const float* __restrict__ x,
                          const float* __restrict__ w,
                          const float* __restrict__ b,
                          OT* __restrict__ out) {
  int pos = blockIdx.x * 256 + threadIdx.x;
  const float4* xp = (const float4*)(x + (size_t)pos * DD);
  float s1 = 0.f, s2 = 0.f;
#pragma unroll
  for (int j = 0; j < 16; ++j) {
    float4 v = xp[j];
    s1 += v.x + v.y + v.z + v.w;
    s2 += v.x * v.x + v.y * v.y + v.z * v.z + v.w * v.w;
  }
  float m  = s1 * (1.f / 64.f);
  float vr = s2 * (1.f / 64.f) - m * m;
  float rs = rsqrtf(vr + 1e-5f);
  OT* op = out + (size_t)pos * DD;
#pragma unroll
  for (int j = 0; j < 16; ++j) {
    float4 v = xp[j];
    int c = j * 4;
    op[c + 0] = (OT)(fmaf((v.x - m) * rs, w[c + 0], b[c + 0]));
    op[c + 1] = (OT)(fmaf((v.y - m) * rs, w[c + 1], b[c + 1]));
    op[c + 2] = (OT)(fmaf((v.z - m) * rs, w[c + 2], b[c + 2]));
    op[c + 3] = (OT)(fmaf((v.w - m) * rs, w[c + 3], b[c + 3]));
  }
}

// ---------------------------------------------------------------------------
// S4D chunked scan.
// wave id -> (b, d, chunk); lane n (lanes 16..31 duplicate states 0..15 so the
// wave stays fully converged for the cross-lane reduce).
// ---------------------------------------------------------------------------
__device__ __forceinline__ void ssm_decay(const float* ldt, const float* lar,
                                          const float* aim, int d, int n,
                                          float& er, float& ei,
                                          float& dtAr, float& dtAi) {
  float dt = __expf(ldt[d]);
  float Ar = -__expf(lar[d * NS + n]);
  float Ai = aim[d * NS + n];
  dtAr = Ar * dt;
  dtAi = Ai * dt;
  float em = __expf(dtAr);
  float sn, cn;
  __sincosf(dtAi, &sn, &cn);
  er = em * cn;
  ei = em * sn;
}

__global__ void scan_p1_kernel(const float* __restrict__ z,
                               const float* __restrict__ ldt,
                               const float* __restrict__ lar,
                               const float* __restrict__ aim,
                               float2* __restrict__ states) {
  int wid   = (blockIdx.x * 256 + threadIdx.x) >> 5;  // (b,d,chunk)
  int lane  = threadIdx.x & 31;
  int chunk = wid & (NCH - 1);
  int d     = (wid >> 6) & (DD - 1);
  int b     = wid >> 12;
  int n     = lane & 15;

  float er, ei, dtAr, dtAi;
  ssm_decay(ldt, lar, aim, d, n, er, ei, dtAr, dtAi);

  const float* zp = z + ((size_t)b * LLEN + (size_t)chunk * CHL) * DD + d;
  float sr = 0.f, si = 0.f;
  for (int t = 0; t < CHL; ++t) {
    float zv = zp[(size_t)t * DD];
    float nr = fmaf(er, sr, fmaf(-ei, si, zv));
    si = fmaf(er, si, ei * sr);
    sr = nr;
  }
  if (lane < 16) states[(size_t)wid * NS + n] = make_float2(sr, si);
}

__global__ void scan_combine_kernel(const float* __restrict__ ldt,
                                    const float* __restrict__ lar,
                                    const float* __restrict__ aim,
                                    const float2* __restrict__ states,
                                    float2* __restrict__ carry) {
  int wid  = (blockIdx.x * 256 + threadIdx.x) >> 5;  // (b,d) : 512 waves
  int lane = threadIdx.x & 31;
  int d    = wid & (DD - 1);
  int n    = lane & 15;

  float er, ei, dtAr, dtAi;
  ssm_decay(ldt, lar, aim, d, n, er, ei, dtAr, dtAi);
  // per-chunk decay e^{dtA * CHL}
  float mag = __expf(dtAr * (float)CHL);
  float sn, cn;
  __sincosf(dtAi * (float)CHL, &sn, &cn);
  float dr = mag * cn, di = mag * sn;

  float cr = 0.f, ci = 0.f;
  size_t base = (size_t)wid * NCH * NS + n;
  for (int c = 0; c < NCH; ++c) {
    if (lane < 16) carry[base + (size_t)c * NS] = make_float2(cr, ci);
    float2 se = states[base + (size_t)c * NS];
    float nr = fmaf(dr, cr, fmaf(-di, ci, se.x));
    ci = fmaf(dr, ci, fmaf(di, cr, se.y));
    cr = nr;
  }
}

__global__ void scan_p2_kernel(const float* __restrict__ z,
                               const float* __restrict__ ldt,
                               const float* __restrict__ lar,
                               const float* __restrict__ aim,
                               const float* __restrict__ cre,
                               const float* __restrict__ cim,
                               const float* __restrict__ dskip,
                               const float2* __restrict__ carry,
                               _Float16* __restrict__ yg) {
  int wid   = (blockIdx.x * 256 + threadIdx.x) >> 5;
  int lane  = threadIdx.x & 31;
  int chunk = wid & (NCH - 1);
  int d     = (wid >> 6) & (DD - 1);
  int b     = wid >> 12;
  int n     = lane & 15;

  float er, ei, dtAr, dtAi;
  ssm_decay(ldt, lar, aim, d, n, er, ei, dtAr, dtAi);
  // dC = (C_re + i C_im) * (e^{dtA} - 1) / A
  float Ar = -__expf(lar[d * NS + n]);
  float Ai = aim[d * NS + n];
  float numr = er - 1.f, numi = ei;
  float inv = 1.f / (Ar * Ar + Ai * Ai);
  float qr = (numr * Ar + numi * Ai) * inv;
  float qi = (numi * Ar - numr * Ai) * inv;
  float Cr = cre[d * NS + n], Ci = cim[d * NS + n];
  float dCr = Cr * qr - Ci * qi;
  float dCi = Cr * qi + Ci * qr;
  float ds = dskip[d];

  float2 cin = carry[(size_t)wid * NS + n];
  float sr = cin.x, si = cin.y;

  const float* zp = z + ((size_t)b * LLEN + (size_t)chunk * CHL) * DD + d;
  _Float16* yp = yg + ((size_t)b * LLEN + (size_t)chunk * CHL) * DD + d;

  for (int t = 0; t < CHL; ++t) {
    float zv = zp[(size_t)t * DD];
    float nr = fmaf(er, sr, fmaf(-ei, si, zv));
    si = fmaf(er, si, ei * sr);
    sr = nr;
    float c = dCr * sr - dCi * si;          // Re(dC * s)
    c += __shfl_xor(c, 1);                  // 16-state reduce (xor stays in half)
    c += __shfl_xor(c, 2);
    c += __shfl_xor(c, 4);
    c += __shfl_xor(c, 8);
    float y = fmaf(ds, zv, 2.f * c);
    float g = gelu_f(y);
    if (lane == 0) yp[(size_t)t * DD] = (_Float16)g;
  }
}

// ---------------------------------------------------------------------------
// WMMA GEMM: C[M,N] = A[M,K] * W[N,K]^T  (+ bias, + epilogue)
//  block = 256 threads = 8 waves; 256 rows per block; wave = 32 rows x N cols
//  (two 16-row M tiles so every LDS B-fragment feeds two WMMAs).
//  EPI 0: GLU + residual into X        (N=128 -> 64)
//  EPI 1: GELU, store f16 Out          (stride N)
//  EPI 2: + bias, residual into X      (N=64)
//  EPI 3: ReLU, store f16 Out          (stride N)
// ---------------------------------------------------------------------------
__device__ __forceinline__ v16h load_frag16(const _Float16* p0,
                                            const _Float16* p1) {
  v8h lo = *(const v8h*)p0;
  v8h hi = *(const v8h*)p1;
  return __builtin_shufflevector(lo, hi, 0, 1, 2, 3, 4, 5, 6, 7, 8, 9, 10, 11,
                                 12, 13, 14, 15);
}

template <int K, int N, int EPI>
__global__ __launch_bounds__(256) void gemm_kernel(
    const _Float16* __restrict__ A, const _Float16* __restrict__ W,
    const float* __restrict__ bias, float* __restrict__ X,
    _Float16* __restrict__ Out) {
  constexpr int NT = N / 16;
  __shared__ __align__(16) _Float16 wlds[N * K];

#if USE_TDM
  // TDM DMA of the whole N*K f16 weight tile into LDS; wave 0 issues (TDM
  // ignores EXEC -> one descriptor), waits TENSORcnt, barrier publishes LDS.
  if (threadIdx.x < 32) {
    tdm_load_tile(W, wlds, (N * K) / 2);
    __builtin_amdgcn_s_wait_tensorcnt(0);
  }
  __syncthreads();
#else
  {
    const uint4* src = (const uint4*)W;
    uint4* dst = (uint4*)wlds;
    constexpr int C4 = (N * K) / 8;
    for (int i = threadIdx.x; i < C4; i += 256) dst[i] = src[i];
  }
  __syncthreads();
#endif

  const int lane = threadIdx.x & 31;
  const int wave = threadIdx.x >> 5;
  const int hl   = lane >> 4;   // which K-half of the fragment this lane owns
  const int nr   = lane & 15;   // row (A) / col (B,C)
  const int rowBase = blockIdx.x * 256 + wave * 32;

  v8f acc0[NT] = {};
  v8f acc1[NT] = {};
  const _Float16* arow0 = A + (size_t)(rowBase + nr) * K;
  const _Float16* arow1 = arow0 + (size_t)16 * K;

  if constexpr (EPI == 0 || EPI == 2) {
    __builtin_prefetch(X + (size_t)(rowBase + nr) * DD, 1, 1);
    __builtin_prefetch(X + (size_t)(rowBase + 16 + nr) * DD, 1, 1);
  }

#pragma unroll
  for (int kk = 0; kk < K / 32; ++kk) {
    // A 16x32 fragment: lane<16 holds K {0..7,16..23}, lane>=16 {8..15,24..31}
    const int ka = kk * 32 + hl * 8;
    v16h a0 = load_frag16(arow0 + ka, arow0 + ka + 16);
    v16h a1 = load_frag16(arow1 + ka, arow1 + ka + 16);
#pragma unroll
    for (int ct = 0; ct < NT; ++ct) {
      // B 32x16 fragment: lane<16 holds K {0..7,16..23}, lane>=16 {8..15,24..31}
      const _Float16* wrow = wlds + (ct * 16 + nr) * K + kk * 32 + hl * 8;
      v16h bf = load_frag16(wrow, wrow + 16);
      acc0[ct] = __builtin_amdgcn_wmma_f32_16x16x32_f16(
          false, a0, false, bf, (short)0, acc0[ct], false, false);
      acc1[ct] = __builtin_amdgcn_wmma_f32_16x16x32_f16(
          false, a1, false, bf, (short)0, acc1[ct], false, false);
    }
  }

  const int mb = hl * 8;  // C/D layout: lane>=16 holds rows M=8..15
  auto epilogue = [&](v8f* acc, int rb) {
    if constexpr (EPI == 0) {
#pragma unroll
      for (int ct = 0; ct < 4; ++ct) {
        const int col = ct * 16 + nr;
        const float ba = bias[col];
        const float bg = bias[col + 64];
#pragma unroll
        for (int r = 0; r < 8; ++r) {
          const size_t pos = (size_t)(rb + mb + r);
          const float av = acc[ct][r] + ba;
          const float gv = acc[ct + 4][r] + bg;
          X[pos * DD + col] += av * sigmoid_f(gv);
        }
      }
    } else if constexpr (EPI == 1) {
#pragma unroll
      for (int ct = 0; ct < NT; ++ct) {
        const int col = ct * 16 + nr;
        const float bc = bias[col];
#pragma unroll
        for (int r = 0; r < 8; ++r) {
          const size_t pos = (size_t)(rb + mb + r);
          Out[pos * N + col] = (_Float16)gelu_f(acc[ct][r] + bc);
        }
      }
    } else if constexpr (EPI == 2) {
#pragma unroll
      for (int ct = 0; ct < NT; ++ct) {
        const int col = ct * 16 + nr;
        const float bc = bias[col];
#pragma unroll
        for (int r = 0; r < 8; ++r) {
          const size_t pos = (size_t)(rb + mb + r);
          X[pos * DD + col] += acc[ct][r] + bc;
        }
      }
    } else {
#pragma unroll
      for (int ct = 0; ct < NT; ++ct) {
        const int col = ct * 16 + nr;
        const float bc = bias[col];
#pragma unroll
        for (int r = 0; r < 8; ++r) {
          const size_t pos = (size_t)(rb + mb + r);
          Out[pos * N + col] = (_Float16)fmaxf(acc[ct][r] + bc, 0.f);
        }
      }
    }
  };
  epilogue(acc0, rowBase);
  epilogue(acc1, rowBase + 16);
}

// ---------------------------------------------------------------------------
// head: out[pos] = w_zero . h[pos,:] + b_zero
// ---------------------------------------------------------------------------
__global__ void head_kernel(const _Float16* __restrict__ h,
                            const float* __restrict__ wz,
                            const float* __restrict__ bz,
                            float* __restrict__ out) {
  int pos = blockIdx.x * 256 + threadIdx.x;
  const _Float16* hp = h + (size_t)pos * DD;
  float s = 0.f;
#pragma unroll
  for (int j = 0; j < DD; ++j) s = fmaf((float)hp[j], wz[j], s);
  out[pos] = s + bz[0];
}

// ---------------------------------------------------------------------------
// launcher
// ---------------------------------------------------------------------------
extern "C" void kernel_launch(void* const* d_in, const int* in_sizes, int n_in,
                              void* d_out, int out_size, void* d_ws,
                              size_t ws_size, hipStream_t stream) {
  (void)in_sizes; (void)n_in; (void)out_size; (void)ws_size;
  const float* in_u      = (const float*)d_in[0];
  const float* w_init    = (const float*)d_in[1];
  const float* b_init    = (const float*)d_in[2];
  const float* s4_ln_w   = (const float*)d_in[3];
  const float* s4_ln_b   = (const float*)d_in[4];
  const float* log_dt    = (const float*)d_in[5];
  const float* log_A_re  = (const float*)d_in[6];
  const float* A_im      = (const float*)d_in[7];
  const float* C_re      = (const float*)d_in[8];
  const float* C_im      = (const float*)d_in[9];
  const float* D_skip    = (const float*)d_in[10];
  const float* s4_out_w  = (const float*)d_in[11];
  const float* s4_out_b  = (const float*)d_in[12];
  const float* ff_ln_w   = (const float*)d_in[13];
  const float* ff_ln_b   = (const float*)d_in[14];
  const float* ff_w1     = (const float*)d_in[15];
  const float* ff_b1     = (const float*)d_in[16];
  const float* ff_w2     = (const float*)d_in[17];
  const float* ff_b2     = (const float*)d_in[18];
  const float* norm_w    = (const float*)d_in[19];
  const float* norm_b    = (const float*)d_in[20];
  const float* w_final1  = (const float*)d_in[21];
  const float* b_final1  = (const float*)d_in[22];
  const float* w_zero    = (const float*)d_in[23];
  const float* b_zero    = (const float*)d_in[24];
  float* out = (float*)d_out;

  // workspace layout (~170.2 MB, buffers aliased across pipeline phases)
  char* base = (char*)d_ws;
  constexpr size_t MB = 1024ull * 1024ull;
  float*    x      = (float*)(base);                 // 64 MB residual stream
  float*    z32    = (float*)(base + 64 * MB);       // 64 MB (alias: h / hfin)
  _Float16* h128   = (_Float16*)(base + 64 * MB);    // alias of z32 region
  _Float16* hfin   = (_Float16*)(base + 64 * MB);    // alias (final head)
  _Float16* yg     = (_Float16*)(base + 128 * MB);   // 32 MB (alias: zh)
  _Float16* zh     = (_Float16*)(base + 128 * MB);   // alias of yg region
  float2*   states = (float2*)(base + 160 * MB);     // 4 MB
  float2*   carry  = (float2*)(base + 165 * MB);     // 4 MB
  _Float16* w_s4   = (_Float16*)(base + 170 * MB);                 // 64 KB
  _Float16* w_f1   = (_Float16*)(base + 170 * MB + 64 * 1024);     // 64 KB
  _Float16* w_f2   = (_Float16*)(base + 170 * MB + 128 * 1024);    // 64 KB
  _Float16* w_fin  = (_Float16*)(base + 170 * MB + 192 * 1024);    // 8 KB

  const dim3 blk(256);
  // --- weight conversion (fp32 -> fp16) ---
  {
    int n = NLAY * 2 * DD * DD;  // 32768
    cvt_f32_f16_kernel<<<(n + 255) / 256, blk, 0, stream>>>(s4_out_w, w_s4, n);
    cvt_f32_f16_kernel<<<(n + 255) / 256, blk, 0, stream>>>(ff_w1, w_f1, n);
    cvt_f32_f16_kernel<<<(n + 255) / 256, blk, 0, stream>>>(ff_w2, w_f2, n);
    int m = DD * DD;
    cvt_f32_f16_kernel<<<(m + 255) / 256, blk, 0, stream>>>(w_final1, w_fin, m);
  }

  // --- init conv + relu ---
  init_kernel<<<(MM * DD) / 256, blk, 0, stream>>>(in_u, w_init, b_init, x);

  const int lnGrid   = MM / 256;                    // 1024
  const int scanGrid = (BB * DD * NCH * 32) / 256;  // 4096
  const int combGrid = (BB * DD * 32) / 256;        // 64
  const int gemmGrid = MM / 256;                    // 1024 (256 rows/block)

  for (int i = 0; i < NLAY; ++i) {
    const float* ldt = log_dt   + (size_t)i * DD;
    const float* lar = log_A_re + (size_t)i * DD * NS;
    const float* aim = A_im     + (size_t)i * DD * NS;

    // S4 block: LN -> scan (S4D + GELU fused) -> WMMA GEMM + GLU + residual
    ln_kernel<float><<<lnGrid, blk, 0, stream>>>(
        x, s4_ln_w + (size_t)i * DD, s4_ln_b + (size_t)i * DD, z32);
    scan_p1_kernel<<<scanGrid, blk, 0, stream>>>(z32, ldt, lar, aim, states);
    scan_combine_kernel<<<combGrid, blk, 0, stream>>>(ldt, lar, aim, states,
                                                      carry);
    scan_p2_kernel<<<scanGrid, blk, 0, stream>>>(
        z32, ldt, lar, aim, C_re + (size_t)i * DD * NS,
        C_im + (size_t)i * DD * NS, D_skip + (size_t)i * DD, carry, yg);
    gemm_kernel<64, 128, 0><<<gemmGrid, blk, 0, stream>>>(
        yg, w_s4 + (size_t)i * 2 * DD * DD, s4_out_b + (size_t)i * 2 * DD, x,
        nullptr);

    // FF block: LN -> WMMA GEMM + GELU -> WMMA GEMM + residual
    ln_kernel<_Float16><<<lnGrid, blk, 0, stream>>>(
        x, ff_ln_w + (size_t)i * DD, ff_ln_b + (size_t)i * DD, zh);
    gemm_kernel<64, 128, 1><<<gemmGrid, blk, 0, stream>>>(
        zh, w_f1 + (size_t)i * 2 * DD * DD, ff_b1 + (size_t)i * 2 * DD, nullptr,
        h128);
    gemm_kernel<128, 64, 2><<<gemmGrid, blk, 0, stream>>>(
        h128, w_f2 + (size_t)i * 2 * DD * DD, ff_b2 + (size_t)i * DD, x,
        nullptr);
  }

  // final LN -> WMMA GEMM + ReLU -> head dot
  ln_kernel<_Float16><<<lnGrid, blk, 0, stream>>>(x, norm_w, norm_b, zh);
  gemm_kernel<64, 64, 3><<<gemmGrid, blk, 0, stream>>>(zh, w_fin, b_final1,
                                                       nullptr, hfin);
  head_kernel<<<lnGrid, blk, 0, stream>>>(hfin, w_zero, b_zero, out);
}
